// VanillaDecoderRNN_62663572849007
// MI455X (gfx1250) — matrix operand order repaired and checked by reading, not compiled, not run on hardware
//
#include <hip/hip_runtime.h>
#include <hip/hip_bf16.h>

#define B_ 64
#define L_ 20
#define E_ 512
#define H_ 512
#define V_ 32000
#define SOS_ 1

typedef __attribute__((ext_vector_type(16))) __bf16 v16bf;
typedef __attribute__((ext_vector_type(8)))  float  v8f;

union AB16 { v16bf v; uint4 u[2]; };

// ---- WMMA fragment loaders (layouts per CDNA5 ISA, wave32) ----

// A: 16x32 bf16 tile of row-major [M][lda] matrix.
// lane<16: row m=lane, elems 0..7 = K[k0+half*8 ..), elems 8..15 = K[k0+16+half*8 ..)
__device__ __forceinline__ v16bf load_a_frag(const __hip_bfloat16* __restrict__ A,
                                             int lda, int row0, int k0, int lane) {
  int m    = lane & 15;
  int half = lane >> 4;                       // 0 or 1 -> K base offset 0 or 8
  const __hip_bfloat16* p = A + (size_t)(row0 + m) * lda + (k0 + half * 8);
  AB16 r;
  r.u[0] = *reinterpret_cast<const uint4*>(p);        // K .. K+8
  r.u[1] = *reinterpret_cast<const uint4*>(p + 16);   // K+16 .. K+24
  return r.v;
}

// B: 32x16 bf16 tile = W^T, W row-major [N][ldw].
// lane<16: col n=n0+lane, elems 0..7 = K[k0+half*8 ..), elems 8..15 = K[k0+16+half*8 ..)
__device__ __forceinline__ v16bf load_b_frag(const __hip_bfloat16* __restrict__ W,
                                             int ldw, int n0, int k0, int lane) {
  int n    = n0 + (lane & 15);
  int koff = k0 + (lane >> 4) * 8;
  const __hip_bfloat16* p = W + (size_t)n * ldw + koff;
  AB16 r;
  r.u[0] = *reinterpret_cast<const uint4*>(p);
  r.u[1] = *reinterpret_cast<const uint4*>(p + 16);
  return r.v;
}

// C[m][n] = sum_k A[m][k] * W[n][k] + bias[n].
// Register-blocked: each wave computes a 64x32 macro-tile (4 M-tiles x 2 N-tiles,
// 8 f32 accumulators). B fragments are reused across the 4 M-tiles, A fragments
// across the 2 N-tiles: 12 b128 loads per 8 WMMA (vs 4 loads/wmma unblocked).
// Block = 256 threads = 8 waves spanning 256 columns of N; grid.y indexes 64-row stripes.
__global__ void wmma_gemm_bias(const __hip_bfloat16* __restrict__ A,
                               const __hip_bfloat16* __restrict__ W,
                               const float* __restrict__ bias,
                               float* __restrict__ C,
                               int K, int lda, int ldw, long ldc) {
  int lane    = threadIdx.x & 31;
  int wave    = threadIdx.x >> 5;
  int n0      = (blockIdx.x * 8 + wave) * 32;   // wave covers columns n0..n0+31
  int rowbase = blockIdx.y * 64;                // ... and rows rowbase..rowbase+63

  v8f acc[4][2] = {};
  for (int k = 0; k < K; k += 32) {
    v16bf b0 = load_b_frag(W, ldw, n0,      k, lane);
    v16bf b1 = load_b_frag(W, ldw, n0 + 16, k, lane);
    #pragma unroll
    for (int t = 0; t < 4; ++t) {
      v16bf a = load_a_frag(A, lda, rowbase + t * 16, k, lane);
      acc[t][0] = __builtin_amdgcn_wmma_f32_16x16x32_bf16(false, a, false, b0,
                                                          (short)0, acc[t][0], false, false);
      acc[t][1] = __builtin_amdgcn_wmma_f32_16x16x32_bf16(false, a, false, b1,
                                                          (short)0, acc[t][1], false, false);
    }
  }

  // C/D layout: VGPR j -> M=j (lanes 0-15) or M=j+8 (lanes 16-31); N = lane&15
  int col0 = n0 + (lane & 15);
  int rofs = (lane >> 4) * 8;
  #pragma unroll
  for (int u = 0; u < 2; ++u) {
    int col  = col0 + u * 16;
    float bv = bias[col];
    #pragma unroll
    for (int t = 0; t < 4; ++t) {
      int r0 = rowbase + t * 16 + rofs;
      #pragma unroll
      for (int j = 0; j < 8; ++j)
        C[(size_t)(r0 + j) * (size_t)ldc + col] = acc[t][u][j] + bv;
    }
  }
}

// ---- precision conversion / gather kernels ----

__global__ void f32_to_bf16(const float* __restrict__ in,
                            __hip_bfloat16* __restrict__ out, int n) {
  int i = blockIdx.x * blockDim.x + threadIdx.x;
  if (i < n) out[i] = __float2bfloat16(in[i]);
}

// x[l*B + b][e] = emb[token(l,b)][e], converted to bf16. token = SOS for l==0 else target[b][l-1].
__global__ void embed_gather_bf16(const float* __restrict__ emb,
                                  const int* __restrict__ target,
                                  __hip_bfloat16* __restrict__ x, int total) {
  int i = blockIdx.x * blockDim.x + threadIdx.x;
  if (i >= total) return;
  int e   = i % E_;
  int row = i / E_;          // row = l*B + b
  int l   = row / B_;
  int b   = row % B_;
  int tok = (l == 0) ? SOS_ : target[b * L_ + (l - 1)];
  x[i] = __float2bfloat16(emb[(size_t)tok * E_ + e]);
}

__global__ void init_h(const float* __restrict__ enc, float* __restrict__ h,
                       __hip_bfloat16* __restrict__ hbf, int n) {
  int i = blockIdx.x * blockDim.x + threadIdx.x;
  if (i < n) { float v = enc[i]; h[i] = v; hbf[i] = __float2bfloat16(v); }
}

// PyTorch GRU cell gate math (gi includes b_ih, gh includes b_hh). Updates h in place.
__global__ void gru_gate(const float* __restrict__ gi, const float* __restrict__ gh,
                         float* __restrict__ h, __hip_bfloat16* __restrict__ hbf, int n) {
  int i = blockIdx.x * blockDim.x + threadIdx.x;
  if (i >= n) return;
  int b = i / H_, j = i % H_;
  const float* gib = gi + (size_t)b * 3 * H_;
  const float* ghb = gh + (size_t)b * 3 * H_;
  float r = 1.0f / (1.0f + __expf(-(gib[j]       + ghb[j])));
  float z = 1.0f / (1.0f + __expf(-(gib[H_ + j]  + ghb[H_ + j])));
  float ng = tanhf(gib[2 * H_ + j] + r * ghb[2 * H_ + j]);
  float hn = (1.0f - z) * ng + z * h[i];
  h[i]   = hn;
  hbf[i] = __float2bfloat16(hn);
}

// In-place log_softmax over rows of length V_. One 256-thread block per row.
__global__ void log_softmax_rows(float* __restrict__ out) {
  __shared__ float red[256];
  float* p = out + (size_t)blockIdx.x * V_;
  int t = threadIdx.x;

  float mx = -INFINITY;
  for (int v = t; v < V_; v += 256) mx = fmaxf(mx, p[v]);
  red[t] = mx; __syncthreads();
  for (int s = 128; s > 0; s >>= 1) { if (t < s) red[t] = fmaxf(red[t], red[t + s]); __syncthreads(); }
  mx = red[0]; __syncthreads();

  float sum = 0.0f;
  for (int v = t; v < V_; v += 256) sum += __expf(p[v] - mx);
  red[t] = sum; __syncthreads();
  for (int s = 128; s > 0; s >>= 1) { if (t < s) red[t] += red[t + s]; __syncthreads(); }
  float lse = __logf(red[0]) + mx;

  for (int v = t; v < V_; v += 256) p[v] = p[v] - lse;
}

__global__ void copy_hidden(const float* __restrict__ h, float* __restrict__ out, int n) {
  int i = blockIdx.x * blockDim.x + threadIdx.x;
  if (i < n) out[i] = h[i];
}

extern "C" void kernel_launch(void* const* d_in, const int* in_sizes, int n_in,
                              void* d_out, int out_size, void* d_ws, size_t ws_size,
                              hipStream_t stream) {
  (void)in_sizes; (void)n_in; (void)out_size; (void)ws_size;
  const float* enc   = (const float*)d_in[0];   // [B,H]
  const int*   tgt   = (const int*)  d_in[1];   // [B,L]
  const float* emb   = (const float*)d_in[2];   // [V,E]
  const float* w_ih  = (const float*)d_in[3];   // [3H,E]
  const float* w_hh  = (const float*)d_in[4];   // [3H,H]
  const float* b_ih  = (const float*)d_in[5];   // [3H]
  const float* b_hh  = (const float*)d_in[6];   // [3H]
  const float* w_out = (const float*)d_in[7];   // [V,H]
  const float* b_out = (const float*)d_in[8];   // [V]
  float* out = (float*)d_out;                   // [B,L,V] log-probs ++ [1,B,H] hidden

  // workspace carve-out (256B aligned)
  char* ws = (char*)d_ws;
  size_t off = 0;
  auto carve = [&](size_t bytes) -> void* {
    void* p = ws + off;
    off = (off + bytes + 255) & ~(size_t)255;
    return p;
  };
  __hip_bfloat16* x_bf    = (__hip_bfloat16*)carve((size_t)L_ * B_ * E_ * 2);   // 1.3 MB
  __hip_bfloat16* wih_bf  = (__hip_bfloat16*)carve((size_t)3 * H_ * E_ * 2);    // 1.6 MB
  __hip_bfloat16* whh_bf  = (__hip_bfloat16*)carve((size_t)3 * H_ * H_ * 2);    // 1.6 MB
  __hip_bfloat16* wout_bf = (__hip_bfloat16*)carve((size_t)V_ * H_ * 2);        // 32.8 MB (L2-resident)
  float*          gi_all  = (float*)carve((size_t)L_ * B_ * 3 * H_ * 4);        // 7.9 MB
  float*          gh      = (float*)carve((size_t)B_ * 3 * H_ * 4);
  float*          hbuf    = (float*)carve((size_t)B_ * H_ * 4);
  __hip_bfloat16* h_bf    = (__hip_bfloat16*)carve((size_t)B_ * H_ * 2);

  int n;
  // one-time bf16 conversions + embedding gather
  n = L_ * B_ * E_;
  embed_gather_bf16<<<(n + 255) / 256, 256, 0, stream>>>(emb, tgt, x_bf, n);
  n = 3 * H_ * E_;
  f32_to_bf16<<<(n + 255) / 256, 256, 0, stream>>>(w_ih, wih_bf, n);
  f32_to_bf16<<<(n + 255) / 256, 256, 0, stream>>>(w_hh, whh_bf, n);
  n = V_ * H_;
  f32_to_bf16<<<(n + 255) / 256, 256, 0, stream>>>(w_out, wout_bf, n);
  n = B_ * H_;
  init_h<<<(n + 255) / 256, 256, 0, stream>>>(enc, hbuf, h_bf, n);

  // gi_all[l*B+b][3H] = x @ w_ih^T + b_ih : M=1280, N=1536, K=512
  {
    dim3 grid((3 * H_) / 256, (L_ * B_) / 64);   // 6 x 20
    wmma_gemm_bias<<<grid, 256, 0, stream>>>(x_bf, wih_bf, b_ih, gi_all,
                                             E_, E_, E_, (long)(3 * H_));
  }

  // sequential GRU steps (stream-ordered dependencies)
  for (int l = 0; l < L_; ++l) {
    {   // gh = h @ w_hh^T + b_hh : M=64, N=1536, K=512
      dim3 grid((3 * H_) / 256, 1);              // 6 x 1
      wmma_gemm_bias<<<grid, 256, 0, stream>>>(h_bf, whh_bf, b_hh, gh,
                                               H_, H_, H_, (long)(3 * H_));
    }
    n = B_ * H_;
    gru_gate<<<(n + 255) / 256, 256, 0, stream>>>(gi_all + (size_t)l * B_ * 3 * H_,
                                                  gh, hbuf, h_bf, n);
    {   // logits[b][l][:] = h_new @ w_out^T + b_out : M=64, N=32000, K=512
        // C base = out + l*V, row (=batch) stride = L*V
      dim3 grid(V_ / 256, 1);                    // 125 x 1
      wmma_gemm_bias<<<grid, 256, 0, stream>>>(h_bf, wout_bf, b_out,
                                               out + (size_t)l * V_,
                                               H_, H_, H_, (long)L_ * V_);
    }
  }

  // in-place log_softmax over all B*L rows, then append final hidden state
  log_softmax_rows<<<B_ * L_, 256, 0, stream>>>(out);
  n = B_ * H_;
  copy_hidden<<<(n + 255) / 256, 256, 0, stream>>>(hbuf, out + (size_t)B_ * L_ * V_, n);
}
